// MOFPointAttention_69870527971623
// MI455X (gfx1250) — matrix operand order repaired and checked by reading, not compiled, not run on hardware
//
#include <hip/hip_runtime.h>
#include <math.h>

typedef __bf16 bf16_t;
typedef __attribute__((ext_vector_type(16))) __bf16 v16bf;
typedef __attribute__((ext_vector_type(8)))  __bf16 bf16x8;
typedef __attribute__((ext_vector_type(8)))  float   v8f;
typedef __attribute__((ext_vector_type(4)))  float   f32x4;

#define MM    3072
#define BBATCH 32
#define NPER  96
#define HH    8
#define CHH   16
#define PQKK  8
#define PVV   12
#define CSS   256
#define CZZ   128
#define EE    (BBATCH*NPER*NPER)   /* 294912 */
#define OF    448                  /* padded o_feats width (422 -> 448) */
#define OUTN  256

// ---------------- WMMA fragment helpers (bf16, 16x16x32) ----------------
static __device__ __forceinline__ v16bf frag_zero() {
  v16bf z;
#pragma unroll
  for (int i = 0; i < 16; ++i) z[i] = (__bf16)0.0f;
  return z;
}
static __device__ __forceinline__ v16bf frag_from_chunks(bf16x8 c0, bf16x8 c1) {
  v16bf a;
#pragma unroll
  for (int i = 0; i < 8; ++i) { a[i] = c0[i]; a[8 + i] = c1[i]; }
  return a;
}
// A (16x32, row-major source). lane holds row (lane&15); halfA = (lane<16)?0:8.
// element e<8 -> K = halfA+e ; e>=8 -> K = 16+halfA+(e-8)  => two contiguous 16B chunks.
static __device__ __forceinline__ v16bf fragA(const bf16_t* rowk, int halfA) {
  bf16x8 c0 = *(const bf16x8*)(rowk + halfA);
  bf16x8 c1 = *(const bf16x8*)(rowk + 16 + halfA);
  return frag_from_chunks(c0, c1);
}
// B (32x16): lane holds column (lane&15); e<8 -> K = halfB+e ; e>=8 -> K = 16+halfB+(e-8).
static __device__ __forceinline__ v16bf fragB(const bf16_t* p) {
  bf16x8 c0 = *(const bf16x8*)(p);
  bf16x8 c1 = *(const bf16x8*)(p + 16);
  return frag_from_chunks(c0, c1);
}
static __device__ __forceinline__ v8f wmma_bf16(v16bf a, v16bf b, v8f c) {
  return __builtin_amdgcn_wmma_f32_16x16x32_bf16(false, a, false, b, (short)0, c, false, false);
}

// ---------------- conversion kernels ----------------
__global__ void k_f32_to_bf16(const float* __restrict__ src, bf16_t* __restrict__ dst, int n) {
  int i = blockIdx.x * blockDim.x + threadIdx.x;
  if (i < n) dst[i] = (bf16_t)src[i];
}

// dst[n][k] (bf16, [Npad][Kpad]) = src[k][n] (f32, [K][N]); zero-padded.
__global__ void k_transpose_pad(const float* __restrict__ src, bf16_t* __restrict__ dst,
                                int K, int N, int Kpad, int Npad) {
  int i = blockIdx.x * blockDim.x + threadIdx.x;
  if (i >= Npad * Kpad) return;
  int n = i / Kpad, k = i - n * Kpad;
  float v = (n < N && k < K) ? src[(size_t)k * N + n] : 0.0f;
  dst[i] = (bf16_t)v;
}

// ---------------- generic WMMA GEMM: C[M,N] = A_bf16[M,K] @ W[K,N] + bias ----------------
// W supplied transposed as WT[N][K] bf16. One 16x16 tile per wave; K % 32 == 0, N % 16 == 0.
__global__ __launch_bounds__(256) void k_gemm_bias(const bf16_t* __restrict__ A,
    const bf16_t* __restrict__ WT, const float* __restrict__ bias,
    float* __restrict__ C, int K, int N, int total_tiles) {
  int wave = threadIdx.x >> 5, lane = threadIdx.x & 31;
  int tile = blockIdx.x * 8 + wave;
  if (tile >= total_tiles) return;
  int tn = N >> 4;
  int ti = tile / tn, tj = tile - ti * tn;
  int l16 = lane & 15;
  int halfA = (lane < 16) ? 0 : 8;
  int halfB = (lane < 16) ? 0 : 8;
  const bf16_t* Arow = A + (size_t)(ti * 16 + l16) * K;
  const bf16_t* Brow = WT + (size_t)(tj * 16 + l16) * K;
  float bv = bias[tj * 16 + l16];
  v8f acc;
#pragma unroll
  for (int r = 0; r < 8; ++r) acc[r] = bv;
  for (int kk = 0; kk < K; kk += 32) {
    v16bf a = fragA(Arow + kk, halfA);
    v16bf b = fragB(Brow + kk + halfB);
    acc = wmma_bf16(a, b, acc);
  }
  int r0 = ti * 16 + ((lane < 16) ? 0 : 8);
  int cn = tj * 16 + l16;
#pragma unroll
  for (int r = 0; r < 8; ++r) C[(size_t)(r0 + r) * N + cn] = acc[r];
}

// ---------------- b_pair = z @ Wb + bb  (z fp32 converted in-register; N padded 8->16) ----
__global__ __launch_bounds__(256) void k_bpair(const float* __restrict__ z,
    const bf16_t* __restrict__ WbT, const float* __restrict__ bb,
    float* __restrict__ bpair, int total_tiles) {
  int wave = threadIdx.x >> 5, lane = threadIdx.x & 31;
  int tile = blockIdx.x * 8 + wave;
  if (tile >= total_tiles) return;
  int l16 = lane & 15;
  int halfA = (lane < 16) ? 0 : 8;
  int halfB = (lane < 16) ? 0 : 8;
  const float* Zrow = z + (size_t)(tile * 16 + l16) * CZZ;
  const bf16_t* Brow = WbT + (size_t)l16 * CZZ;
  float bv = (l16 < 8) ? bb[l16] : 0.0f;
  v8f acc;
#pragma unroll
  for (int r = 0; r < 8; ++r) acc[r] = bv;
  for (int kk = 0; kk < CZZ; kk += 32) {
    f32x4 z0 = *(const f32x4*)(Zrow + kk + halfA);
    f32x4 z1 = *(const f32x4*)(Zrow + kk + halfA + 4);
    f32x4 z2 = *(const f32x4*)(Zrow + kk + 16 + halfA);
    f32x4 z3 = *(const f32x4*)(Zrow + kk + 16 + halfA + 4);
    v16bf a;
#pragma unroll
    for (int i = 0; i < 4; ++i) {
      a[i]      = (__bf16)z0[i];
      a[4 + i]  = (__bf16)z1[i];
      a[8 + i]  = (__bf16)z2[i];
      a[12 + i] = (__bf16)z3[i];
    }
    v16bf b = fragB(Brow + kk + halfB);
    acc = wmma_bf16(a, b, acc);
  }
  if (l16 < 8) {
    int r0 = tile * 16 + ((lane < 16) ? 0 : 8);
#pragma unroll
    for (int r = 0; r < 8; ++r) bpair[(size_t)(r0 + r) * 8 + l16] = acc[r];
  }
}

// ---------------- pack kernel: rotate points, build qhat/khat/vcat + biases ----------------
// qhat[m,h,64]: [0:16)=q*0.125, [16:40)=rot_qpts*hw (p*3+d), [40:64)=0
// khat[m,h,64]: [0:16)=k,       [16:40)=rot_kpts,            [40:64)=0
// vcat[m,h,64]: [0:16)=v,       [16:52)=rot_vpts (p*3+d),    [52:64)=0
// tbias[m,h] = -0.5*hw*|Qpts|^2 ; sbias[m,h] = -0.5*hw*|Kpts|^2 + 0.5*l_nodes
__global__ void k_pack(const float* __restrict__ q_buf, const float* __restrict__ kv_buf,
    const float* __restrict__ qp_raw, const float* __restrict__ kvp_raw,
    const float* __restrict__ rot, const float* __restrict__ trans,
    const float* __restrict__ Lmat, const float* __restrict__ Wl,
    const float* __restrict__ bl, const float* __restrict__ hwv,
    bf16_t* __restrict__ qhat, bf16_t* __restrict__ khat, bf16_t* __restrict__ vcat,
    float* __restrict__ tbias, float* __restrict__ sbias) {
  int idx = blockIdx.x * blockDim.x + threadIdx.x;   // m*8 + h
  if (idx >= MM * HH) return;
  int m = idx >> 3, h = idx & 7;
  int b = m / NPER;
  const float* R = rot + (size_t)m * 9;
  const float* T = trans + (size_t)m * 3;
  float x = hwv[h];
  float sp = (x > 20.0f) ? x : log1pf(expf(x));
  float hw = sp * (1.0f / 12.0f);                    // softplus * sqrt(1/(4*36))
  bf16_t* qh = qhat + (size_t)idx * 64;
  bf16_t* kh = khat + (size_t)idx * 64;
  bf16_t* vc = vcat + (size_t)idx * 64;
#pragma unroll
  for (int c = 0; c < 16; ++c) {
    qh[c] = (bf16_t)(q_buf[(size_t)m * 128 + h * 16 + c] * 0.125f);  // sqrt(1/(4*16))
    kh[c] = (bf16_t)(kv_buf[(size_t)m * 256 + h * 32 + c]);
    vc[c] = (bf16_t)(kv_buf[(size_t)m * 256 + h * 32 + 16 + c]);
  }
  float qn = 0.0f, kn = 0.0f;
#pragma unroll
  for (int p = 0; p < PQKK; ++p) {
    float r0 = qp_raw[(size_t)m * 192 + (0 * 8 + h) * 8 + p];
    float r1 = qp_raw[(size_t)m * 192 + (1 * 8 + h) * 8 + p];
    float r2 = qp_raw[(size_t)m * 192 + (2 * 8 + h) * 8 + p];
#pragma unroll
    for (int d = 0; d < 3; ++d) {
      float v = R[d * 3 + 0] * r0 + R[d * 3 + 1] * r1 + R[d * 3 + 2] * r2 + T[d];
      qn += v * v;
      qh[16 + p * 3 + d] = (bf16_t)(v * hw);
    }
  }
#pragma unroll
  for (int p = 0; p < PQKK + PVV; ++p) {
    float r0 = kvp_raw[(size_t)m * 480 + (0 * 8 + h) * 20 + p];
    float r1 = kvp_raw[(size_t)m * 480 + (1 * 8 + h) * 20 + p];
    float r2 = kvp_raw[(size_t)m * 480 + (2 * 8 + h) * 20 + p];
#pragma unroll
    for (int d = 0; d < 3; ++d) {
      float v = R[d * 3 + 0] * r0 + R[d * 3 + 1] * r1 + R[d * 3 + 2] * r2 + T[d];
      if (p < PQKK) { kn += v * v; kh[16 + p * 3 + d] = (bf16_t)v; }
      else          { vc[16 + (p - PQKK) * 3 + d] = (bf16_t)v; }
    }
  }
#pragma unroll
  for (int c = 40; c < 64; ++c) { qh[c] = (bf16_t)0.0f; kh[c] = (bf16_t)0.0f; }
#pragma unroll
  for (int c = 52; c < 64; ++c) vc[c] = (bf16_t)0.0f;
  tbias[idx] = -0.5f * hw * qn;
  float l = bl[h];
#pragma unroll
  for (int j = 0; j < 6; ++j) l += Lmat[b * 6 + j] * Wl[j * 8 + h];
  sbias[idx] = -0.5f * hw * kn + 0.5f * l;
}

// ---------------- o_feats tail: columns 416..421 = L[batch], 422..447 = 0 ----------------
__global__ void k_ol(const float* __restrict__ Lmat, bf16_t* __restrict__ ofeat) {
  int idx = blockIdx.x * blockDim.x + threadIdx.x;   // m*32 + j
  if (idx >= MM * 32) return;
  int m = idx >> 5, j = idx & 31;
  float v = (j < 6) ? Lmat[(m / NPER) * 6 + j] : 0.0f;
  ofeat[(size_t)m * OF + 416 + j] = (bf16_t)v;
}

// ---------------- fused IPA attention: one block per (batch, head) ----------------
__global__ __launch_bounds__(256) void k_attn(const bf16_t* __restrict__ qhat,
    const bf16_t* __restrict__ khat, const bf16_t* __restrict__ vcat,
    const float* __restrict__ tbias, const float* __restrict__ sbias,
    const float* __restrict__ bpair, bf16_t* __restrict__ ofeat) {
  __shared__ float  Ssm[NPER * NPER];     // logits / exp values (36864 B)
  __shared__ bf16_t Asm[NPER * NPER];     // attn probs bf16     (18432 B)
  __shared__ bf16_t Vt[52 * NPER];        // V^T [c][s] bf16     ( 9984 B)
  int bh = blockIdx.x;
  int b = bh >> 3, h = bh & 7;
  int tid = threadIdx.x, wave = tid >> 5, lane = tid & 31;
  int base = b * NPER;
  int l16 = lane & 15;
  int halfA = (lane < 16) ? 0 : 8;
  int halfB = (lane < 16) ? 0 : 8;

  // stage V^T (52 valid channels: 16 scalar + 36 point coords)
  for (int i = tid; i < 52 * NPER; i += 256) {
    int c = i / NPER, s_ = i - c * NPER;
    Vt[c * NPER + s_] = vcat[((size_t)(base + s_) * HH + h) * 64 + c];
  }

  // S[t][s] = qhat[t] . khat[s] + tbias[t] + sbias[s] + 0.5*bpair ; 6x6 tiles, K=64
  for (int tile = wave; tile < 36; tile += 8) {
    int ti = tile / 6, tj = tile - ti * 6;
    const bf16_t* Arow = qhat + ((size_t)(base + ti * 16 + l16) * HH + h) * 64;
    const bf16_t* Brow = khat + ((size_t)(base + tj * 16 + l16) * HH + h) * 64;
    v8f acc;
#pragma unroll
    for (int r = 0; r < 8; ++r) acc[r] = 0.0f;
#pragma unroll
    for (int kk = 0; kk < 64; kk += 32) {
      v16bf a = fragA(Arow + kk, halfA);
      v16bf bfr = fragB(Brow + kk + halfB);
      acc = wmma_bf16(a, bfr, acc);
    }
    int t0 = ti * 16 + ((lane < 16) ? 0 : 8);
    int s_ = tj * 16 + l16;
    float sb = sbias[(size_t)(base + s_) * HH + h];
#pragma unroll
    for (int r = 0; r < 8; ++r) {
      int t = t0 + r;
      float v = acc[r] + sb + tbias[(size_t)(base + t) * HH + h]
              + 0.5f * bpair[((size_t)b * (NPER * NPER) + (size_t)s_ * NPER + t) * 8 + h];
      Ssm[t * NPER + s_] = v;
    }
  }
  __syncthreads();

  // row softmax over src (each of first 96 threads owns one tgt row)
  if (tid < NPER) {
    int t = tid;
    float mx = -1e30f;
    for (int s_ = 0; s_ < NPER; ++s_) mx = fmaxf(mx, Ssm[t * NPER + s_]);
    float sum = 0.0f;
    for (int s_ = 0; s_ < NPER; ++s_) {
      float e = expf(Ssm[t * NPER + s_] - mx);
      Ssm[t * NPER + s_] = e;
      sum += e;
    }
    float inv = 1.0f / sum;
    for (int s_ = 0; s_ < NPER; ++s_)
      Asm[t * NPER + s_] = (bf16_t)(Ssm[t * NPER + s_] * inv);
  }
  __syncthreads();

  // O[t][c] = attn @ Vcat : 6 x 4 tiles, K = 96 (3 steps)
  for (int tile = wave; tile < 24; tile += 8) {
    int ti = tile / 4, tj = tile - ti * 4;
    int t = ti * 16 + l16;
    int c = tj * 16 + l16;
    v8f acc;
#pragma unroll
    for (int r = 0; r < 8; ++r) acc[r] = 0.0f;
#pragma unroll
    for (int kk = 0; kk < NPER; kk += 32) {
      v16bf a = fragA(&Asm[t * NPER + kk], halfA);
      v16bf bfr = (c < 52) ? fragB(&Vt[c * NPER + kk + halfB]) : frag_zero();
      acc = wmma_bf16(a, bfr, acc);
    }
    if (c < 52) {
      int col;
      if (c < 16) col = h * 16 + c;
      else {
        int pp = c - 16, p = pp / 3, d = pp - p * 3;
        col = 128 + 96 * d + h * PVV + p;
      }
      int t0 = ti * 16 + ((lane < 16) ? 0 : 8);
#pragma unroll
      for (int r = 0; r < 8; ++r)
        ofeat[(size_t)(base + t0 + r) * OF + col] = (bf16_t)acc[r];
    }
  }
}

// ---------------- host-side launch ----------------
extern "C" void kernel_launch(void* const* d_in, const int* in_sizes, int n_in,
                              void* d_out, int out_size, void* d_ws, size_t ws_size,
                              hipStream_t stream) {
  (void)in_sizes; (void)n_in; (void)out_size; (void)ws_size;
  const float* s    = (const float*)d_in[0];
  const float* z    = (const float*)d_in[1];
  const float* rot  = (const float*)d_in[2];
  const float* trn  = (const float*)d_in[3];
  const float* Lm   = (const float*)d_in[4];
  const float* Wq   = (const float*)d_in[5];
  const float* bq   = (const float*)d_in[6];
  const float* Wkv  = (const float*)d_in[7];
  const float* bkv  = (const float*)d_in[8];
  const float* Wqp  = (const float*)d_in[9];
  const float* bqp  = (const float*)d_in[10];
  const float* Wkvp = (const float*)d_in[11];
  const float* bkvp = (const float*)d_in[12];
  const float* Wb   = (const float*)d_in[13];
  const float* bb   = (const float*)d_in[14];
  const float* Wl   = (const float*)d_in[15];
  const float* bl   = (const float*)d_in[16];
  const float* hwv  = (const float*)d_in[17];
  const float* Wout = (const float*)d_in[18];
  const float* bout = (const float*)d_in[19];
  float* out = (float*)d_out;

  char* p = (char*)d_ws;
  auto carve = [&](size_t bytes) -> char* {
    char* r = p;
    p += (bytes + 255) & ~(size_t)255;
    return r;
  };
  bf16_t* s_bf    = (bf16_t*)carve((size_t)MM * CSS * 2);
  bf16_t* WqT     = (bf16_t*)carve((size_t)128 * CSS * 2);
  bf16_t* WkvT    = (bf16_t*)carve((size_t)256 * CSS * 2);
  bf16_t* WqpT    = (bf16_t*)carve((size_t)192 * CSS * 2);
  bf16_t* WkvpT   = (bf16_t*)carve((size_t)480 * CSS * 2);
  bf16_t* WbT     = (bf16_t*)carve((size_t)16 * CZZ * 2);
  bf16_t* WoutT   = (bf16_t*)carve((size_t)OUTN * OF * 2);
  float*  q_buf   = (float*)carve((size_t)MM * 128 * 4);
  float*  kv_buf  = (float*)carve((size_t)MM * 256 * 4);
  float*  qp_raw  = (float*)carve((size_t)MM * 192 * 4);
  float*  kvp_raw = (float*)carve((size_t)MM * 480 * 4);
  float*  bpair   = (float*)carve((size_t)EE * 8 * 4);
  bf16_t* qhat    = (bf16_t*)carve((size_t)MM * HH * 64 * 2);
  bf16_t* khat    = (bf16_t*)carve((size_t)MM * HH * 64 * 2);
  bf16_t* vcat    = (bf16_t*)carve((size_t)MM * HH * 64 * 2);
  float*  tbias   = (float*)carve((size_t)MM * HH * 4);
  float*  sbias   = (float*)carve((size_t)MM * HH * 4);
  bf16_t* ofeat   = (bf16_t*)carve((size_t)MM * OF * 2);

  auto cdiv = [](int a, int b) { return (a + b - 1) / b; };

  // 1) convert s to bf16
  k_f32_to_bf16<<<cdiv(MM * CSS, 256), 256, 0, stream>>>(s, s_bf, MM * CSS);
  // 2) transpose+pad weights to bf16 [N][K]
  k_transpose_pad<<<cdiv(128 * CSS, 256), 256, 0, stream>>>(Wq,   WqT,   CSS, 128, CSS, 128);
  k_transpose_pad<<<cdiv(256 * CSS, 256), 256, 0, stream>>>(Wkv,  WkvT,  CSS, 256, CSS, 256);
  k_transpose_pad<<<cdiv(192 * CSS, 256), 256, 0, stream>>>(Wqp,  WqpT,  CSS, 192, CSS, 192);
  k_transpose_pad<<<cdiv(480 * CSS, 256), 256, 0, stream>>>(Wkvp, WkvpT, CSS, 480, CSS, 480);
  k_transpose_pad<<<cdiv(16 * CZZ, 256), 256, 0, stream>>>(Wb,   WbT,   CZZ, 8,   CZZ, 16);
  k_transpose_pad<<<cdiv(OUTN * OF, 256), 256, 0, stream>>>(Wout, WoutT, 422, OUTN, OF, OUTN);
  // 3) node projections (WMMA GEMMs, K=256)
  {
    int t;
    t = (MM / 16) * (128 / 16);
    k_gemm_bias<<<cdiv(t, 8), 256, 0, stream>>>(s_bf, WqT,   bq,   q_buf,   CSS, 128, t);
    t = (MM / 16) * (256 / 16);
    k_gemm_bias<<<cdiv(t, 8), 256, 0, stream>>>(s_bf, WkvT,  bkv,  kv_buf,  CSS, 256, t);
    t = (MM / 16) * (192 / 16);
    k_gemm_bias<<<cdiv(t, 8), 256, 0, stream>>>(s_bf, WqpT,  bqp,  qp_raw,  CSS, 192, t);
    t = (MM / 16) * (480 / 16);
    k_gemm_bias<<<cdiv(t, 8), 256, 0, stream>>>(s_bf, WkvpT, bkvp, kvp_raw, CSS, 480, t);
  }
  // 4) b_pair = z @ Wb + bb (memory-bound: 151 MB of z)
  {
    int t = EE / 16;
    k_bpair<<<cdiv(t, 8), 256, 0, stream>>>(z, WbT, bb, bpair, t);
  }
  // 5) pack rotated points + prescaled q/k + biases
  k_pack<<<cdiv(MM * HH, 256), 256, 0, stream>>>(q_buf, kv_buf, qp_raw, kvp_raw,
      rot, trn, Lm, Wl, bl, hwv, qhat, khat, vcat, tbias, sbias);
  // 6) o_feats tail columns
  k_ol<<<cdiv(MM * 32, 256), 256, 0, stream>>>(Lm, ofeat);
  // 7) fused attention, one block per (batch, head)
  k_attn<<<BBATCH * HH, 256, 0, stream>>>(qhat, khat, vcat, tbias, sbias, bpair, ofeat);
  // 8) output GEMM: out = o_feats @ Wout + bout  (K padded to 448)
  {
    int t = (MM / 16) * (OUTN / 16);
    k_gemm_bias<<<cdiv(t, 8), 256, 0, stream>>>(ofeat, WoutT, bout, out, OF, OUTN, t);
  }
}
